// EdgeGraspQNet_61409442399000
// MI455X (gfx1250) — matrix operand order repaired and checked by reading, not compiled, not run on hardware
//
#include <hip/hip_runtime.h>

typedef __attribute__((ext_vector_type(16))) _Float16 v16h;
typedef __attribute__((ext_vector_type(8)))  float    v8f;

#define NB   16      // batch
#define NP   2048    // object points
#define KNN  16      // neighbors
#define GRL  128     // gripper points
#define EPSV 1e-5f
#define TM   64      // GEMM M-tile (4 waves x 16 rows)
#define TN   32      // GEMM N-tile (2 x 16 per wave)

// ---------------------------------------------------------------------------
// Brute-force KNN: one thread per query point, insertion-sorted top-16.
// ---------------------------------------------------------------------------
__global__ void knn_kernel(const float* __restrict__ pos, int* __restrict__ idx) {
  int t = blockIdx.x * blockDim.x + threadIdx.x;
  if (t >= NB * NP) return;
  int b = t / NP, n = t % NP;
  const float* pb = pos + (size_t)b * NP * 3;
  float px = pb[n * 3 + 0], py = pb[n * 3 + 1], pz = pb[n * 3 + 2];
  float bd[KNN]; int bi[KNN];
#pragma unroll
  for (int i = 0; i < KNN; ++i) { bd[i] = 3.4e38f; bi[i] = 0; }
  for (int m = 0; m < NP; ++m) {
    float dx = pb[m * 3 + 0] - px, dy = pb[m * 3 + 1] - py, dz = pb[m * 3 + 2] - pz;
    float d = dx * dx + dy * dy + dz * dz;
    if (d < bd[KNN - 1]) {
      int j = KNN - 1;
      while (j > 0 && bd[j - 1] > d) { bd[j] = bd[j - 1]; bi[j] = bi[j - 1]; --j; }
      bd[j] = d; bi[j] = m;
    }
  }
  for (int i = 0; i < KNN; ++i) idx[(size_t)t * KNN + i] = bi[i];
}

// rel[b,n,k,:] = pos[b, idx[b,n,k], :] - pos[b,n,:]
__global__ void rel_kernel(const float* __restrict__ pos, const int* __restrict__ idx,
                           float* __restrict__ rel) {
  int t = blockIdx.x * blockDim.x + threadIdx.x;
  if (t >= NB * NP * KNN) return;
  int bn = t / KNN;
  int b = bn / NP, n = bn % NP;
  const float* pb = pos + (size_t)b * NP * 3;
  int j = idx[t];
  rel[(size_t)t * 3 + 0] = pb[j * 3 + 0] - pb[n * 3 + 0];
  rel[(size_t)t * 3 + 1] = pb[j * 3 + 1] - pb[n * 3 + 1];
  rel[(size_t)t * 3 + 2] = pb[j * 3 + 2] - pb[n * 3 + 2];
}

// E[row] = concat(src[idx[row]][0:Cx], rel[row][0:3]) ; one batch slice.
__global__ void gather_edge_kernel(const float* __restrict__ src, int sld, int Cx,
                                   const int* __restrict__ idxb,
                                   const float* __restrict__ relb,
                                   float* __restrict__ E) {
  int row = blockIdx.x * blockDim.x + threadIdx.x;
  if (row >= NP * KNN) return;
  int j = idxb[row];
  int Ct = Cx + 3;
  float* er = E + (size_t)row * Ct;
  const float* sr = src + (size_t)j * sld;
  for (int c = 0; c < Cx; ++c) er[c] = sr[c];
  er[Cx + 0] = relb[(size_t)row * 3 + 0];
  er[Cx + 1] = relb[(size_t)row * 3 + 1];
  er[Cx + 2] = relb[(size_t)row * 3 + 2];
}

// ---------------------------------------------------------------------------
// Generic WMMA GEMM:  C[M,Nd] = A[M,Kd] @ W[Kd,Nd] (+bias)(+addv[gm>>shift])(+relu)
// f32 inputs converted to f16 and staged into LDS pre-swizzled in fragment
// order (each lane's v16h operand = one 32B-contiguous LDS read, 2x b128).
// Double-buffered LDS with register prefetch: tile k+1 is loaded to registers
// BEFORE the WMMAs of tile k issue, stored to the alternate buffer after, with
// a single __syncthreads per K-step -> global latency hidden behind the XDL.
//
// ISA 7.12.2 swizzles:
//   A(16x32): (M,K) -> lane = (M&15) + (((K>>3)&1)<<4) ; e = (K&7)|((K>>4)<<3)
//   B(32x16): (K,N) -> lane = N + (((K>>3)&1)<<4)      ; e = (K&7)|((K>>4)<<3)
//   C(16x16): vgpr r: lanes 0-15 -> M=r, N=lane ; lanes 16-31 -> M=r+8.
// Block = 128 threads = 4 waves; tile 64x32 (2 WMMAs/wave/K-step, shared A).
// WMMA executes unconditionally (EXEC all-ones); guards only on loads/stores.
// ---------------------------------------------------------------------------
__global__ void __launch_bounds__(128)
wmma_gemm_kernel(const float* __restrict__ A, int lda,
                 const float* __restrict__ W, int ldw,
                 const float* __restrict__ bias,
                 const float* __restrict__ addv, int addshift, int ldadd,
                 float* __restrict__ C, int ldc,
                 int M, int Kd, int Nd, int relu) {
  __shared__ __attribute__((aligned(32))) _Float16 AsL[2][4 * 32 * 16];
  __shared__ __attribute__((aligned(32))) _Float16 BsL[2][2 * 32 * 16];
  const int tid  = threadIdx.x;
  const int wave = tid >> 5, lane = tid & 31;
  const int m0 = blockIdx.x * TM;
  const int n0 = blockIdx.y * TN;

  float ar[4][4];  // A slab chunk per thread (64x32 / 128 thr = 16 floats)
  float br[2][4];  // W tile chunk per thread (32x32 / 128 thr = 8 floats)

  auto load_tileA = [&](int k0) {
    const bool full = (m0 + TM <= M) && (k0 + 32 <= Kd);
#pragma unroll
    for (int j = 0; j < 4; ++j) {
      int i = tid + j * 128;              // 0..511
      int r = i >> 3, c0 = (i & 7) << 2;  // row in slab, K0 within tile
      int gm = m0 + r, gk = k0 + c0;
      if (full) {
        const float* s = A + (size_t)gm * lda + gk;
        ar[j][0] = s[0]; ar[j][1] = s[1]; ar[j][2] = s[2]; ar[j][3] = s[3];
      } else {
#pragma unroll
        for (int q = 0; q < 4; ++q)
          ar[j][q] = (gm < M && gk + q < Kd) ? A[(size_t)gm * lda + gk + q] : 0.0f;
      }
    }
  };
  auto store_tileA = [&](int p) {
#pragma unroll
    for (int j = 0; j < 4; ++j) {
      int i = tid + j * 128;
      int r = i >> 3, c0 = (i & 7) << 2;
      int ln = (r & 15) + (((c0 >> 3) & 1) << 4);
      int e0 = (c0 & 7) | ((c0 >> 4) << 3);
      _Float16* d = &AsL[p][((r >> 4) << 9) + (ln << 4) + e0];
      d[0] = (_Float16)ar[j][0]; d[1] = (_Float16)ar[j][1];
      d[2] = (_Float16)ar[j][2]; d[3] = (_Float16)ar[j][3];
    }
  };
  auto load_tileB = [&](int k0) {
    const bool full = (k0 + 32 <= Kd) && (n0 + TN <= Nd);
#pragma unroll
    for (int j = 0; j < 2; ++j) {
      int i = tid + j * 128;              // 0..255
      int c = i & 31, K0 = (i >> 5) << 2; // N within tile, K0 = 0,4,...,28
      if (full) {
        const float* s = W + (size_t)(k0 + K0) * ldw + n0 + c;
        br[j][0] = s[0];
        br[j][1] = s[(size_t)ldw];
        br[j][2] = s[2 * (size_t)ldw];
        br[j][3] = s[3 * (size_t)ldw];
      } else {
#pragma unroll
        for (int q = 0; q < 4; ++q) {
          int gk = k0 + K0 + q, gn = n0 + c;
          br[j][q] = (gk < Kd && gn < Nd) ? W[(size_t)gk * ldw + gn] : 0.0f;
        }
      }
    }
  };
  auto store_tileB = [&](int p) {
#pragma unroll
    for (int j = 0; j < 2; ++j) {
      int i = tid + j * 128;
      int c = i & 31, K0 = (i >> 5) << 2;
      int ln = (c & 15) + (((K0 >> 3) & 1) << 4);
      int e0 = (K0 & 7) | ((K0 >> 4) << 3);
      _Float16* d = &BsL[p][((c >> 4) << 9) + (ln << 4) + e0];
      d[0] = (_Float16)br[j][0]; d[1] = (_Float16)br[j][1];
      d[2] = (_Float16)br[j][2]; d[3] = (_Float16)br[j][3];
    }
  };

  v8f acc[2] = {};
  // prologue: stage tile 0
  load_tileA(0); load_tileB(0);
  store_tileA(0); store_tileB(0);
  __syncthreads();
  int p = 0;
  for (int k0 = 0; k0 < Kd; k0 += 32) {
    const int k1 = k0 + 32;
    const bool more = (k1 < Kd);
    if (more) {              // register-prefetch next tile before WMMAs issue
      load_tileA(k1);
      load_tileB(k1);
      // L2 prefetch for tile k+2
      if (k1 + 32 < Kd) {
        int gm = m0 + (tid >> 1);
        if (gm < M) __builtin_prefetch(&A[(size_t)gm * lda + k1 + 32 + (tid & 1) * 16], 0, 3);
      }
    }
    // fragments: 32 contiguous bytes per lane -> 2x ds_load_b128 each
    const v16h af  = *(const v16h*)&AsL[p][(wave << 9) + (lane << 4)];
    const v16h bf0 = *(const v16h*)&BsL[p][(lane << 4)];
    const v16h bf1 = *(const v16h*)&BsL[p][512 + (lane << 4)];
    acc[0] = __builtin_amdgcn_wmma_f32_16x16x32_f16(false, af, false, bf0,
                                                    (short)0, acc[0], false, false);
    acc[1] = __builtin_amdgcn_wmma_f32_16x16x32_f16(false, af, false, bf1,
                                                    (short)0, acc[1], false, false);
    if (more) {              // write next tile to alternate buffer
      store_tileA(p ^ 1);
      store_tileB(p ^ 1);
      __syncthreads();
      p ^= 1;
    }
  }
  // epilogue: bias + optional per-batch broadcast add + optional relu
  const int nn = lane & 15;
  const int rb = (lane >> 4) << 3;
#pragma unroll
  for (int h = 0; h < 2; ++h) {
    int gn = n0 + (h << 4) + nn;
    float bv = (bias != nullptr && gn < Nd) ? bias[gn] : 0.0f;
#pragma unroll
    for (int r = 0; r < 8; ++r) {
      int gm = m0 + (wave << 4) + rb + r;
      if (gm < M && gn < Nd) {
        float v = acc[h][r] + bv;
        if (addv != nullptr) v += addv[(size_t)(gm >> addshift) * ldadd + gn];
        if (relu) v = fmaxf(v, 0.0f);
        C[(size_t)gm * ldc + gn] = v;
      }
    }
  }
}

// Per-row GroupNorm over channel groups (matches reference _gn), in place.
__global__ void groupnorm_kernel(float* __restrict__ X, int ld, int R, int Cc, int G,
                                 const float* __restrict__ gamma,
                                 const float* __restrict__ beta, int relu) {
  int t = blockIdx.x * blockDim.x + threadIdx.x;
  if (t >= R * G) return;
  int r = t / G, g = t % G;
  int cs = Cc / G;
  float* x = X + (size_t)r * ld + g * cs;
  float m = 0.f;
  for (int i = 0; i < cs; ++i) m += x[i];
  m /= (float)cs;
  float v = 0.f;
  for (int i = 0; i < cs; ++i) { float d = x[i] - m; v += d * d; }
  v /= (float)cs;
  float inv = rsqrtf(v + EPSV);
  for (int i = 0; i < cs; ++i) {
    float y = (x[i] - m) * inv * gamma[g * cs + i] + beta[g * cs + i];
    x[i] = relu ? fmaxf(y, 0.f) : y;
  }
}

// Gripper GN over (sequence, channel-group) per (batch, group), + relu, in place.
__global__ void gnseq_kernel(float* __restrict__ X, int L, int Cc, int G,
                             const float* __restrict__ gamma,
                             const float* __restrict__ beta) {
  int t = blockIdx.x * blockDim.x + threadIdx.x;
  if (t >= NB * G) return;
  int b = t / G, g = t % G;
  int cs = Cc / G;
  float* base = X + (size_t)b * L * Cc + g * cs;
  float m = 0.f;
  for (int l = 0; l < L; ++l)
    for (int i = 0; i < cs; ++i) m += base[(size_t)l * Cc + i];
  m /= (float)(L * cs);
  float v = 0.f;
  for (int l = 0; l < L; ++l)
    for (int i = 0; i < cs; ++i) { float d = base[(size_t)l * Cc + i] - m; v += d * d; }
  v /= (float)(L * cs);
  float inv = rsqrtf(v + EPSV);
  for (int l = 0; l < L; ++l)
    for (int i = 0; i < cs; ++i) {
      float y = (base[(size_t)l * Cc + i] - m) * inv * gamma[g * cs + i] + beta[g * cs + i];
      base[(size_t)l * Cc + i] = fmaxf(y, 0.f);
    }
}

// dst[n, c (stride dld)] = relu(max_k H[(n*K+k), c])  for one batch slice.
__global__ void kmax_relu_kernel(const float* __restrict__ H, int Cc,
                                 float* __restrict__ dst, int dld) {
  int t = blockIdx.x * blockDim.x + threadIdx.x;
  if (t >= NP * Cc) return;
  int n = t / Cc, c = t % Cc;
  const float* h = H + (size_t)n * KNN * Cc + c;
  float m = h[0];
  for (int k = 1; k < KNN; ++k) m = fmaxf(m, h[(size_t)k * Cc]);
  dst[(size_t)n * dld + c] = fmaxf(m, 0.f);
}

// out[b,c] = max over rows of X[(b*rows+n), c]
__global__ void batchmax_kernel(const float* __restrict__ X, int ld, int rows, int Cc,
                                float* __restrict__ out) {
  int t = blockIdx.x * blockDim.x + threadIdx.x;
  if (t >= NB * Cc) return;
  int b = t / Cc, c = t % Cc;
  const float* x = X + (size_t)b * rows * ld + c;
  float m = x[0];
  for (int n = 1; n < rows; ++n) m = fmaxf(m, x[(size_t)n * ld]);
  out[(size_t)b * Cc + c] = m;
}

__global__ void concat_kernel(const float* __restrict__ a, int Ca,
                              const float* __restrict__ bsrc, int Cb,
                              float* __restrict__ out) {
  int t = blockIdx.x * blockDim.x + threadIdx.x;
  int Ct = Ca + Cb;
  if (t >= NB * Ct) return;
  int b = t / Ct, c = t % Ct;
  out[t] = (c < Ca) ? a[(size_t)b * Ca + c] : bsrc[(size_t)b * Cb + (c - Ca)];
}

// cl_l4: 128 -> 1 with relu
__global__ void final_kernel(const float* __restrict__ x, const float* __restrict__ w,
                             const float* __restrict__ bias, float* __restrict__ out) {
  int b = blockIdx.x * blockDim.x + threadIdx.x;
  if (b >= NB) return;
  float s = bias[0];
  for (int i = 0; i < 128; ++i) s += x[(size_t)b * 128 + i] * w[i];
  out[b] = fmaxf(s, 0.f);
}

// ---------------------------------------------------------------------------
extern "C" void kernel_launch(void* const* d_in, const int* in_sizes, int n_in,
                              void* d_out, int out_size, void* d_ws, size_t ws_size,
                              hipStream_t stream) {
  (void)in_sizes; (void)n_in; (void)out_size; (void)ws_size;
  const float* obj  = (const float*)d_in[0];   // (16,2048,3)
  const float* grip = (const float*)d_in[1];   // (16,128,3)
  auto P = [&](int i) { return (const float*)d_in[i]; };
  // params in setup_inputs dict order; each _lin -> {w,b}, each _gn_p -> {g,b}
  const float *pn1_w1=P(2),  *pn1_b1=P(3),  *pn1_g=P(4),   *pn1_be=P(5),  *pn1_w2=P(6),  *pn1_b2=P(7);
  const float *pn2_w1=P(8),  *pn2_b1=P(9),  *pn2_g=P(10),  *pn2_be=P(11), *pn2_w2=P(12), *pn2_b2=P(13);
  const float *pn3_w1=P(14), *pn3_b1=P(15), *pn3_g=P(16),  *pn3_be=P(17), *pn3_w2=P(18), *pn3_b2=P(19);
  const float *ge1_w1=P(20), *ge1_b1=P(21), *ge1_g1=P(22), *ge1_be1=P(23);
  const float *ge1_w2=P(24), *ge1_b2=P(25), *ge1_g2=P(26), *ge1_be2=P(27);
  const float *ge1_w3=P(28), *ge1_b3=P(29), *ge1_g3=P(30), *ge1_be3=P(31);
  const float *ge2_w1=P(32), *ge2_b1=P(33), *ge2_g1=P(34), *ge2_be1=P(35);
  const float *ge2_w2=P(36), *ge2_b2=P(37);
  const float *gr_w1=P(38),  *gr_b1=P(39),  *gr_g1=P(40),  *gr_be1=P(41);
  const float *gr_w2=P(42),  *gr_b2=P(43),  *gr_g2=P(44),  *gr_be2=P(45);
  const float *gr_w3=P(46),  *gr_b3=P(47),  *gr_g3=P(48),  *gr_be3=P(49);
  const float *cl_w1=P(50),  *cl_b1=P(51),  *cl_g=P(52),   *cl_be=P(53);
  const float *cl_w2=P(54),  *cl_b2=P(55),  *cl_w3=P(56),  *cl_b3=P(57);
  const float *cl_w4=P(58),  *cl_b4=P(59);

  // workspace bump allocator (256B aligned regions)
  char* base = (char*)d_ws;
  size_t off = 0;
  auto alloc = [&](size_t nbytes) -> void* {
    void* p = base + off;
    off += (nbytes + 255) & ~(size_t)255;
    return p;
  };
  int*   idx    = (int*)  alloc((size_t)NB * NP * KNN * 4);
  float* rel    = (float*)alloc((size_t)NB * NP * KNN * 3 * 4);
  float* des    = (float*)alloc((size_t)NB * NP * 224 * 4);       // [h1|h2|h3]
  float* E      = (float*)alloc((size_t)NP * KNN * 68 * 4);       // per-batch edge feats
  float* Hc     = (float*)alloc((size_t)NP * KNN * 128 * 4);      // per-batch hidden
  float* H2     = (float*)alloc((size_t)NP * KNN * 128 * 4);      // per-batch hidden2
  float* ga     = (float*)alloc((size_t)NB * NP * 512 * 4);
  float* gb     = (float*)alloc((size_t)NB * NP * 512 * 4);
  float* pooled = (float*)alloc((size_t)NB * 512 * 4);
  float* tvec   = (float*)alloc((size_t)NB * 1024 * 4);
  float* fa     = (float*)alloc((size_t)NB * NP * 1024 * 4);
  float* fb     = (float*)alloc((size_t)NB * NP * 1024 * 4);
  float* gemd   = (float*)alloc((size_t)NB * 1024 * 4);
  float* g1b    = (float*)alloc((size_t)NB * GRL * 64 * 4);
  float* g2b    = (float*)alloc((size_t)NB * GRL * 128 * 4);
  float* g3b    = (float*)alloc((size_t)NB * GRL * 128 * 4);
  float* gfeat  = (float*)alloc((size_t)NB * 128 * 4);
  float* xcat   = (float*)alloc((size_t)NB * 1152 * 4);
  float* c1     = (float*)alloc((size_t)NB * 512 * 4);
  float* c2     = (float*)alloc((size_t)NB * 256 * 4);
  float* c3     = (float*)alloc((size_t)NB * 128 * 4);

  auto gemm = [&](const float* A, int lda, const float* W, int ldw, const float* bias,
                  const float* addv, int addshift, int ldadd,
                  float* Cm, int ldc, int M, int Kd, int Nd, int relu) {
    dim3 g((unsigned)((M + TM - 1) / TM), (unsigned)((Nd + TN - 1) / TN));
    wmma_gemm_kernel<<<g, 128, 0, stream>>>(A, lda, W, ldw, bias, addv, addshift, ldadd,
                                            Cm, ldc, M, Kd, Nd, relu);
  };

  // ---- KNN graph ----
  knn_kernel<<<(NB * NP + 255) / 256, 256, 0, stream>>>(obj, idx);
  rel_kernel<<<(NB * NP * KNN + 255) / 256, 256, 0, stream>>>(obj, idx, rel);

  // ---- pn edge-conv stack, per batch to bound edge-tensor footprint ----
  const int ER = NP * KNN;  // 32768 edge rows / batch
  for (int b = 0; b < NB; ++b) {
    const int*   idxb = idx + (size_t)b * NP * KNN;
    const float* relb = rel + (size_t)b * NP * KNN * 3;
    float*       desb = des + (size_t)b * NP * 224;
    // pn1: concat(pos_j, rel)=6 -> 32 -> GN8+relu -> 32 -> kmax+relu
    gather_edge_kernel<<<(ER + 255) / 256, 256, 0, stream>>>(obj + (size_t)b * NP * 3, 3, 3, idxb, relb, E);
    gemm(E, 6, pn1_w1, 32, pn1_b1, nullptr, 0, 0, Hc, 32, ER, 6, 32, 0);
    groupnorm_kernel<<<(ER * 8 + 255) / 256, 256, 0, stream>>>(Hc, 32, ER, 32, 8, pn1_g, pn1_be, 1);
    gemm(Hc, 32, pn1_w2, 32, pn1_b2, nullptr, 0, 0, H2, 32, ER, 32, 32, 0);
    kmax_relu_kernel<<<(NP * 32 + 255) / 256, 256, 0, stream>>>(H2, 32, desb + 0, 224);
    // pn2: 35 -> 64 -> GN8+relu -> 64 -> kmax+relu
    gather_edge_kernel<<<(ER + 255) / 256, 256, 0, stream>>>(desb + 0, 224, 32, idxb, relb, E);
    gemm(E, 35, pn2_w1, 64, pn2_b1, nullptr, 0, 0, Hc, 64, ER, 35, 64, 0);
    groupnorm_kernel<<<(ER * 8 + 255) / 256, 256, 0, stream>>>(Hc, 64, ER, 64, 8, pn2_g, pn2_be, 1);
    gemm(Hc, 64, pn2_w2, 64, pn2_b2, nullptr, 0, 0, H2, 64, ER, 64, 64, 0);
    kmax_relu_kernel<<<(NP * 64 + 255) / 256, 256, 0, stream>>>(H2, 64, desb + 32, 224);
    // pn3: 64(+3)=67 -> 128 -> GN16+relu -> 128 -> kmax+relu
    gather_edge_kernel<<<(ER + 255) / 256, 256, 0, stream>>>(desb + 32, 224, 64, idxb, relb, E);
    gemm(E, 67, pn3_w1, 128, pn3_b1, nullptr, 0, 0, Hc, 128, ER, 67, 128, 0);
    groupnorm_kernel<<<(ER * 16 + 255) / 256, 256, 0, stream>>>(Hc, 128, ER, 128, 16, pn3_g, pn3_be, 1);
    gemm(Hc, 128, pn3_w2, 128, pn3_b2, nullptr, 0, 0, H2, 128, ER, 128, 128, 0);
    kmax_relu_kernel<<<(NP * 128 + 255) / 256, 256, 0, stream>>>(H2, 128, desb + 96, 224);
  }

  // ---- ge1 ----
  const int MP = NB * NP;  // 32768 point rows
  gemm(des, 224, ge1_w1, 256, ge1_b1, nullptr, 0, 0, ga, 256, MP, 224, 256, 0);
  groupnorm_kernel<<<(MP * 32 + 255) / 256, 256, 0, stream>>>(ga, 256, MP, 256, 32, ge1_g1, ge1_be1, 1);
  gemm(ga, 256, ge1_w2, 512, ge1_b2, nullptr, 0, 0, gb, 512, MP, 256, 512, 0);
  groupnorm_kernel<<<(MP * 32 + 255) / 256, 256, 0, stream>>>(gb, 512, MP, 512, 32, ge1_g2, ge1_be2, 1);
  gemm(gb, 512, ge1_w3, 512, ge1_b3, nullptr, 0, 0, ga, 512, MP, 512, 512, 0);
  groupnorm_kernel<<<(MP * 32 + 255) / 256, 256, 0, stream>>>(ga, 512, MP, 512, 32, ge1_g3, ge1_be3, 0);
  batchmax_kernel<<<(NB * 512 + 255) / 256, 256, 0, stream>>>(ga, 512, NP, 512, pooled);

  // ---- ge2: split concat(des, pooled)@W into des@W_top + pooled@W_bot ----
  gemm(pooled, 512, ge2_w1 + (size_t)224 * 1024, 1024, ge2_b1, nullptr, 0, 0,
       tvec, 1024, NB, 512, 1024, 0);                             // includes bias
  gemm(des, 224, ge2_w1, 1024, nullptr, tvec, /*gm>>11 = batch*/ 11, 1024,
       fa, 1024, MP, 224, 1024, 0);
  groupnorm_kernel<<<(MP * 32 + 255) / 256, 256, 0, stream>>>(fa, 1024, MP, 1024, 32, ge2_g1, ge2_be1, 1);
  gemm(fa, 1024, ge2_w2, 1024, ge2_b2, nullptr, 0, 0, fb, 1024, MP, 1024, 1024, 0);
  batchmax_kernel<<<(NB * 1024 + 255) / 256, 256, 0, stream>>>(fb, 1024, NP, 1024, gemd);

  // ---- gripper branch ----
  const int MG = NB * GRL;  // 2048 rows
  gemm(grip, 3, gr_w1, 64, gr_b1, nullptr, 0, 0, g1b, 64, MG, 3, 64, 0);
  gnseq_kernel<<<(NB * 8 + 255) / 256, 256, 0, stream>>>(g1b, GRL, 64, 8, gr_g1, gr_be1);
  gemm(g1b, 64, gr_w2, 128, gr_b2, nullptr, 0, 0, g2b, 128, MG, 64, 128, 0);
  gnseq_kernel<<<(NB * 16 + 255) / 256, 256, 0, stream>>>(g2b, GRL, 128, 16, gr_g2, gr_be2);
  gemm(g2b, 128, gr_w3, 128, gr_b3, nullptr, 0, 0, g3b, 128, MG, 128, 128, 0);
  gnseq_kernel<<<(NB * 32 + 255) / 256, 256, 0, stream>>>(g3b, GRL, 128, 32, gr_g3, gr_be3);
  batchmax_kernel<<<(NB * 128 + 255) / 256, 256, 0, stream>>>(g3b, 128, GRL, 128, gfeat);

  // ---- classifier ----
  concat_kernel<<<(NB * 1152 + 255) / 256, 256, 0, stream>>>(gemd, 1024, gfeat, 128, xcat);
  gemm(xcat, 1152, cl_w1, 512, cl_b1, nullptr, 0, 0, c1, 512, NB, 1152, 512, 0);
  groupnorm_kernel<<<(NB * 32 + 255) / 256, 256, 0, stream>>>(c1, 512, NB, 512, 32, cl_g, cl_be, 1);
  gemm(c1, 512, cl_w2, 256, cl_b2, nullptr, 0, 0, c2, 256, NB, 512, 256, 1);
  gemm(c2, 256, cl_w3, 128, cl_b3, nullptr, 0, 0, c3, 128, NB, 256, 128, 1);
  final_kernel<<<1, 32, 0, stream>>>(c3, cl_w4, cl_b4, (float*)d_out);
}
